// Attention_17987323036091
// MI455X (gfx1250) — matrix operand order repaired and checked by reading, not compiled, not run on hardware
//
#include <hip/hip_runtime.h>

typedef __bf16 bf16;
typedef __attribute__((ext_vector_type(16))) __bf16 v16bf;
typedef __attribute__((ext_vector_type(8)))  __bf16 v8bf;
typedef __attribute__((ext_vector_type(8)))  float  v8f;

union V16U { v16bf v; v8bf h[2]; };

// ---------------------------------------------------------------------------
// Geometry constants
//   x: [2, 8, 32, 32, 1152]; windows 4x4x4 -> 256 windows * 64 tokens
#define CDIM   1152
#define C3     3456
#define NTOK   16384      // 256 windows * 64 tokens
#define NWH    4096       // 256 windows * 16 heads
#define HD     72
#define HDQ    96         // hd padded to multiple of 32 (scores K)
#define HDV    80         // hd padded to multiple of 16 (PV N)
#define SCALE  0.11785113019775793f   // 72^-0.5

// window-token index -> flat row in [B,T,H,W] order (also used as gather map)
__device__ __forceinline__ int unpart_row(int t) {
  int r  = t & 63;
  int w  = t >> 6;
  int b  = w & 1;
  int n3 = (w >> 1) & 7;
  int n2 = (w >> 4) & 7;
  int n1 = w >> 7;
  int k3 = r & 3;
  int k2 = (r >> 2) & 3;
  int k1 = r >> 4;
  int T = n1 * 4 + k1, H = n2 * 4 + k2, W = n3 * 4 + k3;
  return ((b * 8 + T) * 32 + H) * 32 + W;
}

// ---------------------------------------------------------------------------
// WMMA tile loaders (wave32, 16-bit element layouts per CDNA5 ISA 7.12.2)
// A tile 16x32: lane L holds row L&15; K elems {k0 + 8*(L>=16) .. +8} and +16.
__device__ __forceinline__ v16bf load_tile_A(const bf16* base, int lda, int row0, int k0) {
  int lane = threadIdx.x & 31;
  const bf16* p = base + (size_t)(row0 + (lane & 15)) * lda + k0 + ((lane >> 4) << 3);
  V16U u;
  u.h[0] = *(const v8bf*)(p);
  u.h[1] = *(const v8bf*)(p + 16);
  return u.v;
}

// B tile 32x16 from row-major [K,N]: lane L holds col n0+(L&15); K rows {k0 + 8*(L>=16) .. +8} and +16.
__device__ __forceinline__ v16bf load_tile_B(const bf16* base, int ldb, int k0, int n0) {
  int lane = threadIdx.x & 31;
  const bf16* p = base + (size_t)(k0 + ((lane >> 4) << 3)) * ldb + n0 + (lane & 15);
  V16U u;
#pragma unroll
  for (int s = 0; s < 8; s++) {
    u.v[s] = p[(size_t)s * ldb];
    u.v[8 + s] = p[(size_t)(16 + s) * ldb];
  }
  return u.v;
}

// ---------------------------------------------------------------------------
// Prep kernels
__global__ void k_gather_x(const float* __restrict__ x, bf16* __restrict__ xw) {
  int idx = blockIdx.x * 256 + threadIdx.x;
  if (idx >= NTOK * CDIM) return;
  int t = idx / CDIM, c = idx - t * CDIM;
  xw[idx] = (bf16)x[(size_t)unpart_row(t) * CDIM + c];
}

__global__ void k_transpose_w(const float* __restrict__ w, bf16* __restrict__ wt, int O, int K) {
  int idx = blockIdx.x * 256 + threadIdx.x;
  if (idx >= O * K) return;
  int o = idx / K, k = idx - o * K;
  wt[(size_t)k * O + o] = (bf16)w[idx];
}

// ---------------------------------------------------------------------------
// Generic bf16 WMMA GEMM: C[M,N] = A[M,K] * Bt[K,N] (+bias), one wave per
// 64x64 output tile. remap=1 scatters rows back to [B,T,H,W] order.
__global__ __launch_bounds__(32)
void k_gemm(const bf16* __restrict__ A, const bf16* __restrict__ Bt,
            float* __restrict__ Cout, const float* __restrict__ bias,
            int K, int N, int remap) {
  int n0 = blockIdx.x * 64;
  int m0 = blockIdx.y * 64;
  int lane = threadIdx.x & 31;

  v8f acc[4][4];
#pragma unroll
  for (int i = 0; i < 4; i++)
#pragma unroll
    for (int j = 0; j < 4; j++)
#pragma unroll
      for (int e = 0; e < 8; e++) acc[i][j][e] = 0.0f;

  for (int k0 = 0; k0 < K; k0 += 32) {
    if (k0 + 32 < K) {  // hint next K-slab into caches
      __builtin_prefetch(A + (size_t)(m0 + (lane & 15)) * K + k0 + 32, 0, 0);
      __builtin_prefetch(Bt + (size_t)(k0 + 32 + lane) * N + n0, 0, 0);
    }
    v16bf a[4];
#pragma unroll
    for (int i = 0; i < 4; i++) a[i] = load_tile_A(A, K, m0 + i * 16, k0);
#pragma unroll
    for (int j = 0; j < 4; j++) {
      v16bf b = load_tile_B(Bt, N, k0, n0 + j * 16);
#pragma unroll
      for (int i = 0; i < 4; i++)
        acc[i][j] = __builtin_amdgcn_wmma_f32_16x16x32_bf16(
            false, a[i], false, b, (short)0, acc[i][j], false, false);
    }
  }

  int lane15 = lane & 15;
  int hgrp = (lane >> 4) << 3;   // lanes 16..31 hold rows M+8
#pragma unroll
  for (int i = 0; i < 4; i++)
#pragma unroll
    for (int j = 0; j < 4; j++)
#pragma unroll
      for (int p = 0; p < 8; p++) {
        int row = m0 + i * 16 + p + hgrp;
        int col = n0 + j * 16 + lane15;
        float val = acc[i][j][p];
        if (bias) val += bias[col];
        int orow = remap ? unpart_row(row) : row;
        Cout[(size_t)orow * N + col] = val;
      }
}

// ---------------------------------------------------------------------------
// RMS-norm (fp32 stats) + pack Q/K/V to bf16 in WMMA-friendly layouts.
// One wave per (window, head, token).
__global__ __launch_bounds__(32)
void k_norm_pack(const float* __restrict__ qkv,
                 const float* __restrict__ qw, const float* __restrict__ kw,
                 bf16* __restrict__ qb, bf16* __restrict__ kb, bf16* __restrict__ vb) {
  int id = blockIdx.x;
  int n = id & 63;
  int mh = id >> 6;                 // m*16 + h
  int lane = threadIdx.x & 31;
  int t = (mh >> 4) * 64 + n;
  const float* qr = qkv + (size_t)t * C3 + (mh & 15) * HD;
  const float* kr = qr + CDIM;
  const float* vr = qr + 2 * CDIM;

  float sq = 0.0f, sk = 0.0f;
  for (int d = lane; d < HD; d += 32) {
    float a = qr[d]; sq += a * a;
    float c = kr[d]; sk += c * c;
  }
#pragma unroll
  for (int m = 16; m >= 1; m >>= 1) {
    sq += __shfl_xor(sq, m, 32);
    sk += __shfl_xor(sk, m, 32);
  }
  float qsc = rsqrtf(sq * (1.0f / HD) + 1e-6f);
  float ksc = rsqrtf(sk * (1.0f / HD) + 1e-6f);

  bf16* qbase = qb + ((size_t)mh * 64 + n) * HDQ;        // [mh][n][96]  (A layout)
  bf16* kbase = kb + (size_t)mh * HDQ * 64;              // [mh][96][64] (B layout, K^T)
  bf16* vbase = vb + ((size_t)mh * 64 + n) * HDV;        // [mh][n][80]  (B layout rows)
  for (int d = lane; d < HDQ; d += 32) {
    float qv = (d < HD) ? qr[d] * qsc * qw[d] : 0.0f;
    float kv = (d < HD) ? kr[d] * ksc * kw[d] : 0.0f;
    qbase[d] = (bf16)qv;
    kbase[(size_t)d * 64 + n] = (bf16)kv;
  }
  for (int d = lane; d < HDV; d += 32)
    vbase[d] = (bf16)((d < HD) ? vr[d] : 0.0f);
}

// ---------------------------------------------------------------------------
// Attention: one wave per (window, head). scores = Q*K^T (K padded to 96),
// fp32 softmax in registers, P staged bf16 in LDS, out = P*V.
__global__ __launch_bounds__(32)
void k_attn(const bf16* __restrict__ qb, const bf16* __restrict__ kb,
            const bf16* __restrict__ vb, bf16* __restrict__ ob) {
  __shared__ bf16 Pl[64 * 64];
  int mh = blockIdx.x;
  int h = mh & 15;
  int m = mh >> 4;
  int lane = threadIdx.x & 31;
  int lane15 = lane & 15;
  int hgrp = (lane >> 4) << 3;

  const bf16* q  = qb + (size_t)mh * 64 * HDQ;
  const bf16* kT = kb + (size_t)mh * HDQ * 64;
  const bf16* v  = vb + (size_t)mh * 64 * HDV;

  v8f acc[4][4];
#pragma unroll
  for (int i = 0; i < 4; i++)
#pragma unroll
    for (int j = 0; j < 4; j++)
#pragma unroll
      for (int e = 0; e < 8; e++) acc[i][j][e] = 0.0f;

  for (int k0 = 0; k0 < HDQ; k0 += 32) {
    v16bf a[4];
#pragma unroll
    for (int i = 0; i < 4; i++) a[i] = load_tile_A(q, HDQ, i * 16, k0);
#pragma unroll
    for (int j = 0; j < 4; j++) {
      v16bf b = load_tile_B(kT, 64, k0, j * 16);
#pragma unroll
      for (int i = 0; i < 4; i++)
        acc[i][j] = __builtin_amdgcn_wmma_f32_16x16x32_bf16(
            false, a[i], false, b, (short)0, acc[i][j], false, false);
    }
  }

  // fp32 softmax: each row lives in one 16-lane half across 4 column tiles.
#pragma unroll
  for (int i = 0; i < 4; i++) {
#pragma unroll
    for (int p = 0; p < 8; p++) {
      float s[4];
#pragma unroll
      for (int j = 0; j < 4; j++) s[j] = acc[i][j][p] * SCALE;
      float mx = fmaxf(fmaxf(s[0], s[1]), fmaxf(s[2], s[3]));
#pragma unroll
      for (int msk = 8; msk >= 1; msk >>= 1) mx = fmaxf(mx, __shfl_xor(mx, msk, 32));
      float e[4], sum = 0.0f;
#pragma unroll
      for (int j = 0; j < 4; j++) { e[j] = __expf(s[j] - mx); sum += e[j]; }
#pragma unroll
      for (int msk = 8; msk >= 1; msk >>= 1) sum += __shfl_xor(sum, msk, 32);
      float inv = 1.0f / sum;
      int row = i * 16 + p + hgrp;
#pragma unroll
      for (int j = 0; j < 4; j++)
        Pl[row * 64 + j * 16 + lane15] = (bf16)(e[j] * inv);
    }
  }
  __syncthreads();

  // out = P[64x64] * V[64x80], write bf16 rows in proj-GEMM A order
#pragma unroll
  for (int i = 0; i < 4; i++) {
    for (int jj = 0; jj < 5; jj++) {
      v8f o;
#pragma unroll
      for (int e = 0; e < 8; e++) o[e] = 0.0f;
#pragma unroll
      for (int kk = 0; kk < 2; kk++) {
        v16bf a = load_tile_A(Pl, 64, i * 16, kk * 32);
        v16bf b = load_tile_B(v, HDV, kk * 32, jj * 16);
        o = __builtin_amdgcn_wmma_f32_16x16x32_bf16(
            false, a, false, b, (short)0, o, false, false);
      }
#pragma unroll
      for (int p = 0; p < 8; p++) {
        int row = i * 16 + p + hgrp;
        int col = jj * 16 + lane15;
        if (col < HD)
          ob[((size_t)m * 64 + row) * CDIM + h * HD + col] = (bf16)o[p];
      }
    }
  }
}

// ---------------------------------------------------------------------------
extern "C" void kernel_launch(void* const* d_in, const int* in_sizes, int n_in,
                              void* d_out, int out_size, void* d_ws, size_t ws_size,
                              hipStream_t stream) {
  const float* x      = (const float*)d_in[0];
  const float* w_qkv  = (const float*)d_in[1];
  const float* qnw    = (const float*)d_in[2];
  const float* knw    = (const float*)d_in[3];
  const float* w_proj = (const float*)d_in[4];
  const float* b_proj = (const float*)d_in[5];
  float* out = (float*)d_out;

  char* p = (char*)d_ws;
  bf16*  xw     = (bf16*)p;  p += (size_t)NTOK * CDIM * 2;     // x windowed, bf16
  bf16*  wqkvT  = (bf16*)p;  p += (size_t)CDIM * C3 * 2;       // [K,N] bf16
  bf16*  wprojT = (bf16*)p;  p += (size_t)CDIM * CDIM * 2;     // [K,N] bf16
  float* qkv    = (float*)p; p += (size_t)NTOK * C3 * 4;       // fp32 qkv
  bf16*  qb     = (bf16*)p;  p += (size_t)NWH * 64 * HDQ * 2;
  bf16*  kb     = (bf16*)p;  p += (size_t)NWH * HDQ * 64 * 2;
  bf16*  vb     = (bf16*)p;  p += (size_t)NWH * 64 * HDV * 2;
  bf16*  ab     = (bf16*)p;  p += (size_t)NTOK * CDIM * 2;     // attention out, bf16

  k_gather_x<<<(NTOK * CDIM + 255) / 256, 256, 0, stream>>>(x, xw);
  k_transpose_w<<<(C3 * CDIM + 255) / 256, 256, 0, stream>>>(w_qkv, wqkvT, C3, CDIM);
  k_transpose_w<<<(CDIM * CDIM + 255) / 256, 256, 0, stream>>>(w_proj, wprojT, CDIM, CDIM);

  // QKV GEMM: [16384,1152] x [1152,3456]
  k_gemm<<<dim3(C3 / 64, NTOK / 64), 32, 0, stream>>>(xw, wqkvT, qkv, nullptr, CDIM, C3, 0);

  // RMS norm + pack
  k_norm_pack<<<NWH * 64, 32, 0, stream>>>(qkv, qnw, knw, qb, kb, vb);

  // windowed attention
  k_attn<<<NWH, 32, 0, stream>>>(qb, kb, vb, ab);

  // Proj GEMM + bias + window-unpartition scatter: [16384,1152] x [1152,1152]
  k_gemm<<<dim3(CDIM / 64, NTOK / 64), 32, 0, stream>>>(ab, wprojT, out, b_proj, CDIM, CDIM, 1);
}
